// Block_25477746000258
// MI455X (gfx1250) — matrix operand order repaired and checked by reading, not compiled, not run on hardware
//
#include <hip/hip_runtime.h>
#include <hip/hip_bf16.h>

// ---------------------------------------------------------------------------
// Transformer block (RMSNorm -> MHA -> residual -> RMSNorm -> SwiGLU -> residual)
// for MI455X (gfx1250, wave32, WMMA + TDM + async-LDS loads).
// f16 data path, f32 accumulation.
// ---------------------------------------------------------------------------

typedef __attribute__((ext_vector_type(16))) _Float16 v16h;
typedef __attribute__((ext_vector_type(8)))  _Float16 v8h;
typedef __attribute__((ext_vector_type(8)))  float    v8f;
typedef __attribute__((ext_vector_type(4)))  unsigned int v4u;
typedef __attribute__((ext_vector_type(8)))  int v8i;
typedef __attribute__((ext_vector_type(4)))  int v4i;

#define WMMA_F16(a, b, c) \
  __builtin_amdgcn_wmma_f32_16x16x32_f16(false, (a), false, (b), (short)0, (c), false, false)

// Model constants
static constexpr int D_MODEL  = 1024;
static constexpr int N_HEADS  = 16;
static constexpr int HEAD_DIM = 64;
static constexpr int D_HIDDEN = 4096;
static constexpr int SEQ      = 2048;
static constexpr int BATCH    = 2;
static constexpr int M_TOK    = BATCH * SEQ;   // 4096 token rows

// ---------------------------------------------------------------------------
// LDS byte offset of a __shared__ object (generic -> AS(3) -> integer)
// ---------------------------------------------------------------------------
__device__ inline unsigned lds_off(const void* p) {
  return (unsigned)(unsigned long long)
      (__attribute__((address_space(3))) const void*)p;
}

// ---------------------------------------------------------------------------
// Async global->LDS load (128-bit per lane), tracked with ASYNCcnt.
// ---------------------------------------------------------------------------
__device__ inline void async_load_b128(unsigned lds_byte, const void* g) {
  asm volatile("global_load_async_to_lds_b128 %0, %1, off"
               :: "v"(lds_byte), "v"(g) : "memory");
}
__device__ inline void wait_asynccnt0() {
  asm volatile("s_wait_asynccnt 0" ::: "memory");
}

// ---------------------------------------------------------------------------
// Tensor Data Mover: DMA a 128-row x 32-half 2D tile (f16) from global to LDS.
// LDS rows are padded 32 data halfs + 8 pad halfs (stride 40 halfs):
//   pad_interval code 3 = 16 DWORDs (64B) of data, pad_amount code 3 = 4 DWORDs.
// Descriptor per CDNA5 D# group0/group1 layout; groups 2/3 zero (2D tensor).
// ---------------------------------------------------------------------------
__device__ inline void tdm_load_tile(unsigned lds_byte, const _Float16* g,
                                     int row_stride_halfs) {
  unsigned long long ga = (unsigned long long)g;
  v4u g0;
  g0[0] = 1u;                                           // count=1, user descriptor
  g0[1] = lds_byte;                                     // lds_addr
  g0[2] = (unsigned)ga;                                 // global_addr[31:0]
  g0[3] = (unsigned)((ga >> 32) & 0x01FFFFFFu) | (2u << 30);  // addr[56:32] | type=2
  v8i g1;
  // workgroup_mask=0 | data_size=1(2B)<<16 | pad_enable<<20 | pad_interval=3<<22
  // | pad_amount=3<<25
  g1[0] = (1 << 16) | (1 << 20) | (3 << 22) | (3 << 25);
  unsigned td0 = (unsigned)row_stride_halfs;            // tensor_dim0 (elems/row)
  unsigned td1 = 128u;                                  // tensor_dim1 (rows)
  g1[1] = (int)((td0 & 0xffffu) << 16);                 // tensor_dim0[15:0]
  g1[2] = (int)((td0 >> 16) | ((td1 & 0xffffu) << 16)); // dim0[31:16] | dim1[15:0]
  g1[3] = (int)((td1 >> 16) | (32u << 16));             // dim1[31:16] | tile_dim0=32
  g1[4] = 128;                                          // tile_dim1=128, tile_dim2=0
  g1[5] = row_stride_halfs;                             // tensor_dim0_stride[31:0]
  g1[6] = 0;                                            // stride[47:32] | dim1_stride lo
  g1[7] = 0;
  v4i z4 = {0, 0, 0, 0};
#if defined(__clang_major__) && __clang_major__ >= 23
  v8i z8 = {0, 0, 0, 0, 0, 0, 0, 0};
  __builtin_amdgcn_tensor_load_to_lds(g0, g1, z4, z4, z8, 0);
#else
  __builtin_amdgcn_tensor_load_to_lds(g0, g1, z4, z4, 0);
#endif
}

// ---------------------------------------------------------------------------
// Fragment loaders matching CDNA5 WMMA VGPR layouts (ISA 7.12.2).
// ---------------------------------------------------------------------------
__device__ inline v16h load_frag_a(const _Float16* base, int stride) {
  int lane = threadIdx.x & 31;
  int m = lane & 15, h = lane >> 4;
  const _Float16* row = base + m * stride;
  union { v16h v; v8h p[2]; } u;
  u.p[0] = *(const v8h*)(row + h * 8);
  u.p[1] = *(const v8h*)(row + 16 + h * 8);
  return u.v;
}

__device__ inline v16h load_frag_bt(const _Float16* base, int stride) {
  int lane = threadIdx.x & 31;
  int n = lane & 15, h = lane >> 4;
  const _Float16* row = base + n * stride;
  union { v16h v; v8h p[2]; } u;
  u.p[0] = *(const v8h*)(row + h * 8);
  u.p[1] = *(const v8h*)(row + 16 + h * 8);
  return u.v;
}

// ---------------------------------------------------------------------------
// Weight transpose + cast: in (R x C) f32 row-major -> out (C x R) f16 row-major
// ---------------------------------------------------------------------------
__global__ __launch_bounds__(256)
void transpose_cast_kernel(const float* __restrict__ in, _Float16* __restrict__ out,
                           int R, int C) {
  __shared__ float tile[32][33];
  int r0 = blockIdx.y * 32, c0 = blockIdx.x * 32;
  int lr = threadIdx.x >> 5;   // 0..7
  int lc = threadIdx.x & 31;   // 0..31
#pragma unroll
  for (int i = 0; i < 4; ++i) {
    int r = lr + i * 8;
    tile[r][lc] = in[(size_t)(r0 + r) * C + c0 + lc];
  }
  __syncthreads();
#pragma unroll
  for (int i = 0; i < 4; ++i) {
    int r = lr + i * 8;  // row offset in output space
    out[(size_t)(c0 + r) * R + r0 + lc] = (_Float16)tile[lc][r];
  }
}

// ---------------------------------------------------------------------------
// RMSNorm over D_MODEL + cast to f16. One block per token row.
// ---------------------------------------------------------------------------
__global__ __launch_bounds__(256)
void rmsnorm_cast_kernel(const float* __restrict__ x, const float* __restrict__ w,
                         _Float16* __restrict__ out) {
  __shared__ float red[256];
  int row = blockIdx.x;
  const float* xr = x + (size_t)row * D_MODEL;
  float ss = 0.f;
  for (int c = threadIdx.x; c < D_MODEL; c += 256) {
    float v = xr[c];
    ss += v * v;
  }
  red[threadIdx.x] = ss;
  __syncthreads();
  for (int s = 128; s > 0; s >>= 1) {
    if (threadIdx.x < s) red[threadIdx.x] += red[threadIdx.x + s];
    __syncthreads();
  }
  float inv = rsqrtf(red[0] / (float)D_MODEL + 1e-6f);
  for (int c = threadIdx.x; c < D_MODEL; c += 256)
    out[(size_t)row * D_MODEL + c] = (_Float16)(w[c] * xr[c] * inv);
}

// ---------------------------------------------------------------------------
// WMMA GEMM:  C(MxN) = A(MxK) @ B(KxN), with B supplied transposed (Bt: N x K).
// 128x128 tile per block, 256 threads = 8 waves, each wave 32(M) x 64(N).
// Double-buffered LDS; tiles staged by the Tensor Data Mover (wave 0),
// overlapped with WMMA on the current buffer.
// Epilogues: 0 = f16 store, 1 = silu->f16, 2 = (*mul)->f16, 3 = f32 bias+resid.
// ---------------------------------------------------------------------------
template <int EPI>
__global__ __launch_bounds__(256)
void gemm_bt_kernel(const _Float16* __restrict__ A, const _Float16* __restrict__ Bt,
                    const float* __restrict__ bias, const float* __restrict__ resid,
                    const _Float16* __restrict__ mul,
                    _Float16* __restrict__ outH, float* __restrict__ outF,
                    int M, int N, int K) {
  __shared__ _Float16 sA[2][128 * 40];
  __shared__ _Float16 sB[2][128 * 40];

  const int tile_m = blockIdx.y * 128;
  const int tile_n = blockIdx.x * 128;
  const int tid = threadIdx.x;
  const int w = tid >> 5;
  const int lane = tid & 31;
  const int wm = (w & 3) * 32;   // 4 waves along M, 2 m-tiles each
  const int wn = (w >> 2) * 64;  // 2 waves along N, 4 n-tiles each

  const _Float16* Abase = A + (size_t)tile_m * K;
  const _Float16* Bbase = Bt + (size_t)tile_n * K;

  // Prologue: TDM-stage the first K-chunk into buffer 0.
  if (w == 0) {
    tdm_load_tile(lds_off(&sA[0][0]), Abase, K);
    tdm_load_tile(lds_off(&sB[0][0]), Bbase, K);
    __builtin_amdgcn_s_wait_tensorcnt(0);
  }
  __syncthreads();

  v8f acc[2][4] = {};
  int cur = 0;
  for (int k0 = 0; k0 < K; k0 += 32) {
    const int nxt = cur ^ 1;
    const bool more = (k0 + 32) < K;
    if (w == 0 && more) {
      tdm_load_tile(lds_off(&sA[nxt][0]), Abase + (k0 + 32), K);
      tdm_load_tile(lds_off(&sB[nxt][0]), Bbase + (k0 + 32), K);
    }

    v16h af[2], bf[4];
#pragma unroll
    for (int i = 0; i < 2; ++i)
      af[i] = load_frag_a(&sA[cur][(wm + i * 16) * 40], 40);
#pragma unroll
    for (int j = 0; j < 4; ++j)
      bf[j] = load_frag_bt(&sB[cur][(wn + j * 16) * 40], 40);
#pragma unroll
    for (int i = 0; i < 2; ++i)
#pragma unroll
      for (int j = 0; j < 4; ++j)
        acc[i][j] = WMMA_F16(af[i], bf[j], acc[i][j]);

    if (w == 0 && more) __builtin_amdgcn_s_wait_tensorcnt(0);
    __syncthreads();
    cur = nxt;
  }

  // Epilogue: D element r -> (m = 8*(lane/16)+r, n = lane%16) within 16x16 tile
  const int ln = lane & 15, lh = lane >> 4;
#pragma unroll
  for (int i = 0; i < 2; ++i) {
#pragma unroll
    for (int j = 0; j < 4; ++j) {
#pragma unroll
      for (int r = 0; r < 8; ++r) {
        int m = tile_m + wm + i * 16 + lh * 8 + r;
        int n = tile_n + wn + j * 16 + ln;
        float v = acc[i][j][r];
        size_t idx = (size_t)m * N + n;
        if (EPI == 0) {
          outH[idx] = (_Float16)v;
        } else if (EPI == 1) {
          outH[idx] = (_Float16)(v / (1.f + __expf(-v)));  // silu
        } else if (EPI == 2) {
          outH[idx] = (_Float16)(v * (float)mul[idx]);
        } else {
          outF[idx] = resid[idx] + v + bias[n];
        }
      }
    }
  }
}

// ---------------------------------------------------------------------------
// Fused flash attention. Block = (q-tile of 128 rows) x head x batch.
// K/V streamed in tiles of 64; K tile staged via async global->LDS loads.
// S = Q K^T via WMMA (f32, LDS); online softmax; O += P V via WMMA.
// Q/K/V buffers laid out [token(4096)][head*64+hd] f16.
// ---------------------------------------------------------------------------
__global__ __launch_bounds__(256)
void flash_attn_kernel(const _Float16* __restrict__ qb, const _Float16* __restrict__ kb,
                       const _Float16* __restrict__ vb, _Float16* __restrict__ ob) {
  __shared__ _Float16 sK[64 * 72];    // key-major tile: [key][hd]
  __shared__ _Float16 sVt[64 * 72];   // transposed V tile: [hd][key]
  __shared__ float    sS[128 * 68];   // score tile f32
  __shared__ _Float16 sP[128 * 72];   // softmax'd probs f16
  __shared__ float    sO[128 * 64];   // output accumulator
  __shared__ float    sM[128], sL[128];

  const int b = blockIdx.z;
  const int head = blockIdx.y;
  const int q0 = blockIdx.x * 128;
  const float inv_scale = 0.125f;  // 1/sqrt(64)

  const int tid = threadIdx.x;
  const int w = tid >> 5;
  const int lane = tid & 31;
  const int wm = (w & 3) * 32;   // 32 query rows per wave (2 m-tiles)
  const int wn = (w >> 2) * 32;  // 32 key cols per wave (2 n-tiles)

  for (int i = tid; i < 128 * 64; i += 256) sO[i] = 0.f;
  if (tid < 128) { sM[tid] = -1e30f; sL[tid] = 0.f; }

  // Register-resident Q fragments: 2 m-tiles x 2 k-chunks (hd = 64)
  v16h qf[2][2];
  {
    int m = lane & 15, h = lane >> 4;
#pragma unroll
    for (int i = 0; i < 2; ++i) {
      const _Float16* row =
          qb + (size_t)(b * SEQ + q0 + wm + i * 16 + m) * D_MODEL + head * HEAD_DIM;
#pragma unroll
      for (int c = 0; c < 2; ++c) {
        union { v16h v; v8h p[2]; } u;
        u.p[0] = *(const v8h*)(row + c * 32 + h * 8);
        u.p[1] = *(const v8h*)(row + c * 32 + 16 + h * 8);
        qf[i][c] = u.v;
      }
    }
  }
  __syncthreads();

  for (int kt = 0; kt < SEQ; kt += 64) {
    // Stage K tile (64 keys x 64 hd) via async global->LDS B128 loads
    for (int q = tid; q < 64 * 8; q += 256) {
      int r = q >> 3, c8 = (q & 7) * 8;
      async_load_b128(lds_off(&sK[r * 72 + c8]),
                      &kb[(size_t)(b * SEQ + kt + r) * D_MODEL + head * HEAD_DIM + c8]);
    }
    // Stage V^T tile (64 hd x 64 keys)
    for (int q = tid; q < 64 * 64; q += 256) {
      int c = q & 63, n = q >> 6;  // c = hd, n = key
      sVt[c * 72 + n] = vb[(size_t)(b * SEQ + kt + n) * D_MODEL + head * HEAD_DIM + c];
    }
    wait_asynccnt0();
    __syncthreads();

    // S = Q K^T / scale  (wave: 32 rows x 32 keys)
#pragma unroll
    for (int i = 0; i < 2; ++i) {
#pragma unroll
      for (int j = 0; j < 2; ++j) {
        v8f d = {};
#pragma unroll
        for (int c = 0; c < 2; ++c) {
          v16h bfr = load_frag_bt(&sK[(wn + j * 16) * 72 + c * 32], 72);
          d = WMMA_F16(qf[i][c], bfr, d);
        }
        int ln = lane & 15, lh = lane >> 4;
        int mrow = wm + i * 16 + lh * 8;
#pragma unroll
        for (int r = 0; r < 8; ++r)
          sS[(mrow + r) * 68 + wn + j * 16 + ln] = d[r] * inv_scale;
      }
    }
    __syncthreads();

    // Online softmax (one thread per query row)
    if (tid < 128) {
      int m = tid;
      float mi = sM[m], li = sL[m];
      float mt = -1e30f;
      for (int c = 0; c < 64; ++c) mt = fmaxf(mt, sS[m * 68 + c]);
      float mnew = fmaxf(mi, mt);
      float alpha = __expf(mi - mnew);
      float sum = 0.f;
      for (int c = 0; c < 64; ++c) {
        float p = __expf(sS[m * 68 + c] - mnew);
        sum += p;
        sP[m * 72 + c] = (_Float16)p;
      }
      sM[m] = mnew;
      sL[m] = li * alpha + sum;
      for (int c = 0; c < 64; ++c) sO[m * 64 + c] *= alpha;
    }
    __syncthreads();

    // O += P @ V  (A = sP 128x64, Bt = sVt: hd x key)
#pragma unroll
    for (int i = 0; i < 2; ++i) {
#pragma unroll
      for (int j = 0; j < 2; ++j) {
        v8f d = {};
#pragma unroll
        for (int c = 0; c < 2; ++c) {
          v16h a = load_frag_a(&sP[(wm + i * 16) * 72 + c * 32], 72);
          v16h bfr = load_frag_bt(&sVt[(wn + j * 16) * 72 + c * 32], 72);
          d = WMMA_F16(a, bfr, d);
        }
        int ln = lane & 15, lh = lane >> 4;
        int mrow = wm + i * 16 + lh * 8;
#pragma unroll
        for (int r = 0; r < 8; ++r)
          sO[(mrow + r) * 64 + wn + j * 16 + ln] += d[r];
      }
    }
    __syncthreads();
  }

  // Normalize and write head output (concat-head layout)
  for (int q = tid; q < 128 * 64; q += 256) {
    int m = q >> 6, c = q & 63;
    ob[(size_t)(b * SEQ + q0 + m) * D_MODEL + head * HEAD_DIM + c] =
        (_Float16)(sO[m * 64 + c] / sL[m]);
  }
}

// ---------------------------------------------------------------------------
// Host-side orchestration
// ---------------------------------------------------------------------------
extern "C" void kernel_launch(void* const* d_in, const int* in_sizes, int n_in,
                              void* d_out, int out_size, void* d_ws, size_t ws_size,
                              hipStream_t stream) {
  const float* x      = (const float*)d_in[0];
  const float* W_q    = (const float*)d_in[1];
  const float* W_k    = (const float*)d_in[2];
  const float* W_v    = (const float*)d_in[3];
  const float* W_o    = (const float*)d_in[4];
  const float* b_o    = (const float*)d_in[5];
  const float* anw    = (const float*)d_in[6];
  const float* fnw    = (const float*)d_in[7];
  const float* W_gate = (const float*)d_in[8];
  const float* W_hid  = (const float*)d_in[9];
  const float* W_out  = (const float*)d_in[10];
  const float* b_out  = (const float*)d_in[11];
  float* out = (float*)d_out;

  // Workspace layout (bump allocator, 256B aligned)
  size_t off = 0;
  auto alloc = [&](size_t bytes) -> void* {
    void* p = (char*)d_ws + off;
    off += (bytes + 255) & ~(size_t)255;
    return p;
  };
  const size_t DD  = (size_t)D_MODEL * D_MODEL;
  const size_t DH  = (size_t)D_MODEL * D_HIDDEN;
  const size_t MD  = (size_t)M_TOK * D_MODEL;
  const size_t MH  = (size_t)M_TOK * D_HIDDEN;

  _Float16* Wq_t   = (_Float16*)alloc(DD * 2);
  _Float16* Wk_t   = (_Float16*)alloc(DD * 2);
  _Float16* Wv_t   = (_Float16*)alloc(DD * 2);
  _Float16* Wo_t   = (_Float16*)alloc(DD * 2);
  _Float16* Wg_t   = (_Float16*)alloc(DH * 2);
  _Float16* Wh_t   = (_Float16*)alloc(DH * 2);
  _Float16* Wout_t = (_Float16*)alloc(DH * 2);
  _Float16* xn     = (_Float16*)alloc(MD * 2);
  _Float16* qbuf   = (_Float16*)alloc(MD * 2);
  _Float16* kbuf   = (_Float16*)alloc(MD * 2);
  _Float16* vbuf   = (_Float16*)alloc(MD * 2);
  _Float16* abuf   = (_Float16*)alloc(MD * 2);
  float*    x1     = (float*)   alloc(MD * 4);
  _Float16* gbuf   = (_Float16*)alloc(MH * 2);
  _Float16* hgbuf  = (_Float16*)alloc(MH * 2);
  (void)ws_size; (void)in_sizes; (void)n_in; (void)out_size;

  dim3 blk(256);

  // 1. Cast + transpose weights to f16 (N x K layout)
  transpose_cast_kernel<<<dim3(D_MODEL / 32, D_MODEL / 32), blk, 0, stream>>>(W_q, Wq_t, D_MODEL, D_MODEL);
  transpose_cast_kernel<<<dim3(D_MODEL / 32, D_MODEL / 32), blk, 0, stream>>>(W_k, Wk_t, D_MODEL, D_MODEL);
  transpose_cast_kernel<<<dim3(D_MODEL / 32, D_MODEL / 32), blk, 0, stream>>>(W_v, Wv_t, D_MODEL, D_MODEL);
  transpose_cast_kernel<<<dim3(D_MODEL / 32, D_MODEL / 32), blk, 0, stream>>>(W_o, Wo_t, D_MODEL, D_MODEL);
  transpose_cast_kernel<<<dim3(D_HIDDEN / 32, D_MODEL / 32), blk, 0, stream>>>(W_gate, Wg_t, D_MODEL, D_HIDDEN);
  transpose_cast_kernel<<<dim3(D_HIDDEN / 32, D_MODEL / 32), blk, 0, stream>>>(W_hid, Wh_t, D_MODEL, D_HIDDEN);
  transpose_cast_kernel<<<dim3(D_MODEL / 32, D_HIDDEN / 32), blk, 0, stream>>>(W_out, Wout_t, D_HIDDEN, D_MODEL);

  // 2. attn-RMSNorm(x) -> xn (f16)
  rmsnorm_cast_kernel<<<M_TOK, blk, 0, stream>>>(x, anw, xn);

  // 3. Q, K, V projections (f16 out)
  dim3 gProj(D_MODEL / 128, M_TOK / 128);
  gemm_bt_kernel<0><<<gProj, blk, 0, stream>>>(xn, Wq_t, nullptr, nullptr, nullptr,
                                               qbuf, nullptr, M_TOK, D_MODEL, D_MODEL);
  gemm_bt_kernel<0><<<gProj, blk, 0, stream>>>(xn, Wk_t, nullptr, nullptr, nullptr,
                                               kbuf, nullptr, M_TOK, D_MODEL, D_MODEL);
  gemm_bt_kernel<0><<<gProj, blk, 0, stream>>>(xn, Wv_t, nullptr, nullptr, nullptr,
                                               vbuf, nullptr, M_TOK, D_MODEL, D_MODEL);

  // 4. Flash attention per (q-tile, head, batch)
  flash_attn_kernel<<<dim3(SEQ / 128, N_HEADS, BATCH), blk, 0, stream>>>(qbuf, kbuf, vbuf, abuf);

  // 5. Output projection + bias + residual -> x1 (f32)
  gemm_bt_kernel<3><<<gProj, blk, 0, stream>>>(abuf, Wo_t, b_o, x, nullptr,
                                               nullptr, x1, M_TOK, D_MODEL, D_MODEL);

  // 6. ffn-RMSNorm(x1) -> xn (f16, reuse buffer)
  rmsnorm_cast_kernel<<<M_TOK, blk, 0, stream>>>(x1, fnw, xn);

  // 7. gate = silu(xn @ W_gate) (f16)
  dim3 gFfn(D_HIDDEN / 128, M_TOK / 128);
  gemm_bt_kernel<1><<<gFfn, blk, 0, stream>>>(xn, Wg_t, nullptr, nullptr, nullptr,
                                              gbuf, nullptr, M_TOK, D_HIDDEN, D_MODEL);

  // 8. hg = (xn @ W_hidden) * gate (f16)
  gemm_bt_kernel<2><<<gFfn, blk, 0, stream>>>(xn, Wh_t, nullptr, nullptr, gbuf,
                                              hgbuf, nullptr, M_TOK, D_HIDDEN, D_MODEL);

  // 9. out = x1 + hg @ W_out + b_out (f32)
  gemm_bt_kernel<3><<<gProj, blk, 0, stream>>>(hgbuf, Wout_t, b_out, x1, nullptr,
                                               nullptr, out, M_TOK, D_MODEL, D_HIDDEN);
}
